// GroupedQueryAttention_18305150616141
// MI455X (gfx1250) — matrix operand order repaired and checked by reading, not compiled, not run on hardware
//
#include <hip/hip_runtime.h>
#include <hip/hip_bf16.h>
#include <stdint.h>

// ---------------- CDNA5 WMMA types ----------------
typedef __bf16 bf16_t;
typedef __attribute__((ext_vector_type(16))) bf16_t v16bf;
typedef __attribute__((ext_vector_type(8)))  float  v8f;
typedef __attribute__((ext_vector_type(4)))  unsigned int v4u;
typedef __attribute__((ext_vector_type(8)))  int v8i;
typedef __attribute__((ext_vector_type(4)))  int v4i;

// Problem constants (match reference)
#define PB   2
#define PS   2048
#define PE   2048
#define PH   16
#define PHKV 4
#define PD   128
#define PG   4      // PH / PHKV

#if defined(__has_builtin)
#if __has_builtin(__builtin_amdgcn_tensor_load_to_lds) && __has_builtin(__builtin_amdgcn_s_wait_tensorcnt)
#define USE_TDM 1
#endif
#endif

__device__ __forceinline__ unsigned short f2bf(float f) {
  unsigned u = __builtin_bit_cast(unsigned, f);
  u += 0x7fffu + ((u >> 16) & 1u);           // round to nearest even
  return (unsigned short)(u >> 16);
}

struct Pair128 { uint4 a, b; };

// Build a 16-element bf16 fragment from two 16B chunks (global or LDS).
__device__ __forceinline__ v16bf ld2(const unsigned short* p0, const unsigned short* p1) {
  Pair128 c;
  c.a = *reinterpret_cast<const uint4*>(p0);
  c.b = *reinterpret_cast<const uint4*>(p1);
  return __builtin_bit_cast(v16bf, c);
}

__device__ __forceinline__ v8f wmma_bf16(v16bf a, v16bf b, v8f c) {
  // D = A(16x32) * B(32x16) + C, f32 accumulate
  return __builtin_amdgcn_wmma_f32_16x16x32_bf16(false, a, false, b, (short)0, c, false, false);
}

#ifdef USE_TDM
// Issue a 2-D TDM tile load: tile_d1 rows x tile_d0 elements (2-byte elems),
// global row stride = stride0 elements, packed contiguously into LDS at lds_off.
// D# bit layout per CDNA5 ISA 8.3/8.4 (group0: count/lds/global/type; group1:
// data_size + tensor dims + tile dims + dim0 stride).
__device__ __forceinline__ void tdm_load_2d(unsigned lds_off, const void* gptr,
                                            unsigned tile_d0, unsigned tile_d1,
                                            unsigned tensor_d0, unsigned tensor_d1,
                                            unsigned long long stride0_elems) {
  unsigned long long ga = (unsigned long long)(uintptr_t)gptr;
  v4u g0;
  g0[0] = 1u;                                   // count=1, user mode
  g0[1] = lds_off;                              // LDS byte address
  g0[2] = (unsigned)ga;                         // global addr [31:0]
  g0[3] = (unsigned)(ga >> 32) | (2u << 30);    // global addr [56:32] | type=2
  v8i g1;
  g1[0] = (int)(1u << 16);                      // workgroup_mask=0, data_size=1 (2B)
  g1[1] = (int)((tensor_d0 & 0xFFFFu) << 16);   // [47:32]=0, tensor_dim0 lo16
  g1[2] = (int)((tensor_d0 >> 16) | ((tensor_d1 & 0xFFFFu) << 16));
  g1[3] = (int)((tensor_d1 >> 16) | (tile_d0 << 16));   // tile_dim0
  g1[4] = (int)(tile_d1 & 0xFFFFu);             // tile_dim1, tile_dim2=0
  g1[5] = (int)(unsigned)(stride0_elems & 0xFFFFFFFFull);
  g1[6] = (int)(unsigned)((stride0_elems >> 32) & 0xFFFFull);  // dim1_stride=0
  g1[7] = 0;
  v4i z4 = {0, 0, 0, 0};
#if __clang_major__ >= 23
  v8i z8 = {0, 0, 0, 0, 0, 0, 0, 0};
  __builtin_amdgcn_tensor_load_to_lds(g0, g1, z4, z4, z8, 0);
#else
  __builtin_amdgcn_tensor_load_to_lds(g0, g1, z4, z4, 0);
#endif
}
#endif

// ---------------- f32 -> bf16 convert ----------------
__global__ __launch_bounds__(256)
void cvt_f32_bf16(const float* __restrict__ in, unsigned short* __restrict__ out, int n) {
  int i = blockIdx.x * 256 + threadIdx.x;
  if (i < n) out[i] = f2bf(in[i]);
}

// ---------------- Generic GEMM: C[M,N] = A[M,K] * W[N,K]^T ----------------
// A, W row-major bf16 (K contiguous). 256 threads = 8 waves.
// Block tile 128x128; wave tile 32x64 (2x4 WMMA frags).
// MODE 0: bf16 row-major out; MODE 1: f32 row-major; MODE 2: bf16 transposed (B,HKV,D,S)
template<int MODE>
__global__ __launch_bounds__(256)
void gemm_bf16(const unsigned short* __restrict__ A,
               const unsigned short* __restrict__ W,
               void* __restrict__ Cout,
               int M, int N, int K,
               int S_, int D_, int HKV_)
{
  const int lane = threadIdx.x & 31;
  const int wid  = threadIdx.x >> 5;
  const int wrow = wid & 3;
  const int wcol = wid >> 2;
  const int row0 = blockIdx.x * 128 + wrow * 32;
  const int col0 = blockIdx.y * 128 + wcol * 64;

  const int mA  = lane & 15;
  const int hiA = lane >> 4;
  const int kbA = hiA * 8;
  const int kbB = hiA * 8;

  v8f acc[2][4];
#pragma unroll
  for (int i = 0; i < 2; ++i)
#pragma unroll
    for (int j = 0; j < 4; ++j) acc[i][j] = {};

  for (int k0 = 0; k0 < K; k0 += 32) {
    v16bf af[2], bfv[4];
#pragma unroll
    for (int i = 0; i < 2; ++i) {
      const unsigned short* p = A + (size_t)(row0 + i * 16 + mA) * K + k0 + kbA;
      __builtin_prefetch(p + 32, 0, 1);
      af[i] = ld2(p, p + 16);
    }
#pragma unroll
    for (int j = 0; j < 4; ++j) {
      const unsigned short* p = W + (size_t)(col0 + j * 16 + mA) * K + k0 + kbB;
      __builtin_prefetch(p + 32, 0, 1);
      bfv[j] = ld2(p, p + 16);
    }
#pragma unroll
    for (int i = 0; i < 2; ++i)
#pragma unroll
      for (int j = 0; j < 4; ++j)
        acc[i][j] = wmma_bf16(af[i], bfv[j], acc[i][j]);
  }

#pragma unroll
  for (int i = 0; i < 2; ++i)
#pragma unroll
    for (int j = 0; j < 4; ++j)
#pragma unroll
      for (int r = 0; r < 8; ++r) {
        int row = row0 + i * 16 + r + 8 * hiA;
        int col = col0 + j * 16 + mA;
        float v = acc[i][j][r];
        if (MODE == 0) {
          ((unsigned short*)Cout)[(size_t)row * N + col] = f2bf(v);
        } else if (MODE == 1) {
          ((float*)Cout)[(size_t)row * N + col] = v;
        } else {
          int bb = row / S_, ss = row % S_;
          int hh = col / D_, dd = col % D_;
          ((unsigned short*)Cout)[(((size_t)bb * HKV_ + hh) * D_ + dd) * (size_t)S_ + ss] = f2bf(v);
        }
      }
}

// ---------------- Causal flash attention (GQA), block-cooperative ----------------
// Q: bf16 (B*S, H*D). K: bf16 (B*S, HKV*D). Vt: bf16 (B,HKV,D,S). O: bf16 (B*S, H*D).
// 8 waves/block, each owns a 16-row Q tile; all waves iterate a uniform causal range so
// K/V tiles can be staged once per block in LDS (TDM double-buffered), cutting K/V L2
// traffic 8x versus per-wave streaming.
__global__ __launch_bounds__(256)
void gqa_attn(const unsigned short* __restrict__ Qb,
              const unsigned short* __restrict__ Kb,
              const unsigned short* __restrict__ Vt,
              unsigned short* __restrict__ O)
{
  __shared__ unsigned short Kt[2][32][128];   // [buf][key][d]   8KB each
  __shared__ unsigned short Vw[2][128][32];   // [buf][d][key]   8KB each
  __shared__ unsigned short Pt[8][16 * 32];   // per-wave P tile

  const int lane = threadIdx.x & 31;
  const int wid  = threadIdx.x >> 5;
  const int tid  = threadIdx.x;

  int bid = blockIdx.x;
  const int qblk = bid & 15;  bid >>= 4;       // S/128 = 16
  const int hq   = bid & 15;  bid >>= 4;       // H = 16
  const int b    = bid;                        // B = 2
  const int hk   = hq >> 2;                    // hq / G
  const int qtile = qblk * 8 + wid;            // 0..127

  const int n   = lane & 15;
  const int hi  = lane >> 4;
  const int kbA = hi * 8;
  const int kbB = hi * 8;

  // Resident Q fragments: 16 rows x 128 d, 4 A-frags over d
  v16bf qf[4];
  {
    const unsigned short* qrow =
        Qb + ((size_t)(b * PS + qtile * 16 + n)) * (PH * PD) + hq * PD + kbA;
#pragma unroll
    for (int ds = 0; ds < 4; ++ds) {
      const unsigned short* p = qrow + ds * 32;
      qf[ds] = ld2(p, p + 16);
    }
  }

  v8f od[8];
#pragma unroll
  for (int dc = 0; dc < 8; ++dc) od[dc] = {};
  float mrun[8], lrun[8];
#pragma unroll
  for (int r = 0; r < 8; ++r) { mrun[r] = -1e30f; lrun[r] = 0.0f; }

  const float scale = 0.08838834764831845f;    // 1/sqrt(128)
  const int npair = 4 * qblk + 4;              // uniform across block (max qtile = 8*qblk+7)

  const unsigned short* Kbase = Kb + (size_t)(b * PS) * (PHKV * PD) + hk * PD;
  const unsigned short* Vbase = Vt + ((size_t)(b * PHKV + hk)) * PD * (size_t)PS;

#ifdef USE_TDM
  // Stage step 0 via Tensor Data Mover (one wave issues; EXEC-independent DMA).
  if (wid == 0) {
    tdm_load_2d((unsigned)(uintptr_t)&Kt[0][0][0], Kbase, /*d0*/128, /*d1*/32, 128, 32, PHKV * PD);
    tdm_load_2d((unsigned)(uintptr_t)&Vw[0][0][0], Vbase, /*d0*/32, /*d1*/128, 32, 128, PS);
  }
#else
  // Fallback: cooperative load+ds_store staging of step 0.
  {
#pragma unroll
    for (int c = tid; c < 512; c += 256) {          // K tile: 32 rows x 16 chunks
      int row = c >> 4, col = (c & 15) * 8;
      *(uint4*)&Kt[0][row][col] = *(const uint4*)(Kbase + (size_t)row * (PHKV * PD) + col);
    }
#pragma unroll
    for (int c = tid; c < 512; c += 256) {          // V tile: 128 rows x 4 chunks
      int row = c >> 2, col = (c & 3) * 8;
      *(uint4*)&Vw[0][row][col] = *(const uint4*)(Vbase + (size_t)row * PS + col);
    }
  }
#endif

  for (int t = 0; t < npair; ++t) {
    const int nb = t & 1;
    const int tile0 = 2 * t, tile1 = 2 * t + 1;

    // ---- prefetch step t+1 into the other buffer, then publish buffer t ----
#ifdef USE_TDM
    if (wid == 0) {
      if (t + 1 < npair) {
        const unsigned short* kg = Kbase + (size_t)(t + 1) * 32 * (PHKV * PD);
        const unsigned short* vg = Vbase + (size_t)(t + 1) * 32;
        tdm_load_2d((unsigned)(uintptr_t)&Kt[nb ^ 1][0][0], kg, 128, 32, 128, 32, PHKV * PD);
        tdm_load_2d((unsigned)(uintptr_t)&Vw[nb ^ 1][0][0], vg, 32, 128, 32, 128, PS);
        __builtin_amdgcn_s_wait_tensorcnt(2);   // step t's two DMAs done (in-order)
      } else {
        __builtin_amdgcn_s_wait_tensorcnt(0);
      }
    }
#else
    if (t + 1 < npair) {
      const unsigned short* kg = Kbase + (size_t)(t + 1) * 32 * (PHKV * PD);
      const unsigned short* vg = Vbase + (size_t)(t + 1) * 32;
#pragma unroll
      for (int c = tid; c < 512; c += 256) {
        int row = c >> 4, col = (c & 15) * 8;
        *(uint4*)&Kt[nb ^ 1][row][col] = *(const uint4*)(kg + (size_t)row * (PHKV * PD) + col);
      }
#pragma unroll
      for (int c = tid; c < 512; c += 256) {
        int row = c >> 2, col = (c & 3) * 8;
        *(uint4*)&Vw[nb ^ 1][row][col] = *(const uint4*)(vg + (size_t)row * PS + col);
      }
    }
    asm volatile("s_wait_dscnt 0" ::: "memory");
#endif
    __syncthreads();   // buffer nb is now valid for every wave

    // ---- scores: S(16x32) = Q(16x128) * K(128x32)^T from LDS, 8 WMMAs ----
    v8f sc0 = {}, sc1 = {};
#pragma unroll
    for (int ds = 0; ds < 4; ++ds) {
      const unsigned short* p0 = &Kt[nb][n][ds * 32 + kbB];
      v16bf kf0 = ld2(p0, p0 + 16);
      sc0 = wmma_bf16(qf[ds], kf0, sc0);
      const unsigned short* p1 = &Kt[nb][16 + n][ds * 32 + kbB];
      v16bf kf1 = ld2(p1, p1 + 16);
      sc1 = wmma_bf16(qf[ds], kf1, sc1);
    }

    // ---- online softmax (row = r + 8*hi; col/key = n) ----
    float e0v[8], e1v[8], scf[8];
#pragma unroll
    for (int r = 0; r < 8; ++r) {
      const int rowq = qtile * 16 + r + 8 * hi;
      float s0 = sc0[r] * scale;
      if (tile0 * 16 + n > rowq) s0 = -1e30f;
      float s1 = sc1[r] * scale;
      if (tile1 * 16 + n > rowq) s1 = -1e30f;
      float rmax = fmaxf(s0, s1);
#pragma unroll
      for (int off = 1; off < 16; off <<= 1)
        rmax = fmaxf(rmax, __shfl_xor(rmax, off, 32));
      const float mnew = fmaxf(mrun[r], rmax);
      const float sf = __expf(mrun[r] - mnew);
      mrun[r] = mnew;
      const float e0 = __expf(s0 - mnew);
      const float e1 = __expf(s1 - mnew);
      float ps = e0 + e1;
#pragma unroll
      for (int off = 1; off < 16; off <<= 1)
        ps += __shfl_xor(ps, off, 32);
      lrun[r] = lrun[r] * sf + ps;
      scf[r] = sf; e0v[r] = e0; e1v[r] = e1;
    }

#pragma unroll
    for (int dc = 0; dc < 8; ++dc)
#pragma unroll
      for (int r = 0; r < 8; ++r) od[dc][r] *= scf[r];

    // ---- P round-trip through per-wave LDS tile: C-layout -> A-layout ----
#pragma unroll
    for (int r = 0; r < 8; ++r) {
      const int rr = r + 8 * hi;
      Pt[wid][rr * 32 + n]      = f2bf(e0v[r]);
      Pt[wid][rr * 32 + 16 + n] = f2bf(e1v[r]);
    }
    asm volatile("s_wait_dscnt 0" ::: "memory");   // wave-local visibility
    const unsigned short* pp = &Pt[wid][n * 32 + kbA];
    v16bf pf = ld2(pp, pp + 16);
    asm volatile("" ::: "memory");

    // ---- O(16x128) += P(16x32) * V(32x128) from LDS V^T tile, 8 WMMAs ----
#pragma unroll
    for (int dc = 0; dc < 8; ++dc) {
      const unsigned short* vp = &Vw[nb][dc * 16 + n][kbB];
      v16bf vf = ld2(vp, vp + 16);
      od[dc] = wmma_bf16(pf, vf, od[dc]);
    }

    __syncthreads();   // all reads of buffer nb done before it is overwritten at t+2
  }

  // ---- normalize and store O (bf16, row-major B*S x H*D) ----
  float invl[8];
#pragma unroll
  for (int r = 0; r < 8; ++r) invl[r] = 1.0f / lrun[r];
#pragma unroll
  for (int dc = 0; dc < 8; ++dc)
#pragma unroll
    for (int r = 0; r < 8; ++r) {
      const int rowq = qtile * 16 + r + 8 * hi;
      O[((size_t)(b * PS + rowq)) * (PH * PD) + hq * PD + dc * 16 + n] =
          f2bf(od[dc][r] * invl[r]);
    }
}

// ---------------- host-side orchestration ----------------
extern "C" void kernel_launch(void* const* d_in, const int* in_sizes, int n_in,
                              void* d_out, int out_size, void* d_ws, size_t ws_size,
                              hipStream_t stream) {
  (void)in_sizes; (void)n_in; (void)out_size; (void)ws_size;
  const float* query = (const float*)d_in[0];
  const float* key   = (const float*)d_in[1];
  const float* value = (const float*)d_in[2];
  // d_in[3] = attn_mask (causal tril) — hardcoded in the attention kernel
  const float* Wq = (const float*)d_in[4];
  const float* Wk = (const float*)d_in[5];
  const float* Wv = (const float*)d_in[6];
  const float* Wo = (const float*)d_in[7];

  const int BS  = PB * PS;        // 4096
  const int HD  = PH * PD;        // 2048
  const int KVD = PHKV * PD;      // 512

  size_t off = 0;
  auto wsalloc = [&](size_t bytes) -> unsigned short* {
    unsigned short* p = (unsigned short*)((char*)d_ws + off);
    off += (bytes + 255) & ~(size_t)255;
    return p;
  };
  unsigned short* Xq  = wsalloc((size_t)BS * PE * 2);
  unsigned short* Xk  = wsalloc((size_t)BS * PE * 2);
  unsigned short* Xv  = wsalloc((size_t)BS * PE * 2);
  unsigned short* Wqb = wsalloc((size_t)HD * PE * 2);
  unsigned short* Wkb = wsalloc((size_t)KVD * PE * 2);
  unsigned short* Wvb = wsalloc((size_t)KVD * PE * 2);
  unsigned short* Wob = wsalloc((size_t)PE * HD * 2);
  unsigned short* Qb  = wsalloc((size_t)BS * HD * 2);
  unsigned short* Kb  = wsalloc((size_t)BS * KVD * 2);
  unsigned short* Vtb = wsalloc((size_t)BS * KVD * 2);
  unsigned short* Ob  = wsalloc((size_t)BS * HD * 2);

  auto cvt = [&](const float* src, unsigned short* dst, int n) {
    cvt_f32_bf16<<<(n + 255) / 256, 256, 0, stream>>>(src, dst, n);
  };
  cvt(query, Xq, BS * PE);
  cvt(key,   Xk, BS * PE);
  cvt(value, Xv, BS * PE);
  cvt(Wq, Wqb, HD * PE);
  cvt(Wk, Wkb, KVD * PE);
  cvt(Wv, Wvb, KVD * PE);
  cvt(Wo, Wob, PE * HD);

  // Projections
  gemm_bf16<0><<<dim3(BS / 128, HD / 128),  256, 0, stream>>>(Xq, Wqb, Qb,  BS, HD,  PE, 0, 0, 0);
  gemm_bf16<0><<<dim3(BS / 128, KVD / 128), 256, 0, stream>>>(Xk, Wkb, Kb,  BS, KVD, PE, 0, 0, 0);
  gemm_bf16<2><<<dim3(BS / 128, KVD / 128), 256, 0, stream>>>(Xv, Wvb, Vtb, BS, KVD, PE, PS, PD, PHKV);

  // Attention (TDM-staged, double-buffered K/V in LDS)
  gqa_attn<<<PB * PH * (PS / 128), 256, 0, stream>>>(Qb, Kb, Vtb, Ob);

  // Output projection -> f32
  gemm_bf16<1><<<dim3(BS / 128, PE / 128), 256, 0, stream>>>(Ob, Wob, d_out, BS, PE, HD, 0, 0, 0);
}
